// TopKastLinear_41429254537837
// MI455X (gfx1250) — compile-verified, matrix operand order checked by reading, not run on hardware
//
#include <hip/hip_runtime.h>

// Problem sizes (match reference)
#define BSZ   2048
#define IN_F  4096
#define OUT_F 4096
#define NW    (OUT_F * IN_F)          // 16777216 weights
#define K0    15099493                // floor(0.9 * (NW-1)); quantile frac = 0.5

// GEMM tiling
#define BM   128
#define BN   128
#define BKK  32
#define LDSS 40                       // LDS row stride in bf16 elems (80B, 16B-aligned)

typedef __attribute__((ext_vector_type(16))) __bf16 v16bf;
typedef __attribute__((ext_vector_type(8)))  __bf16 bf16x8;
typedef __attribute__((ext_vector_type(8)))  float  v8f;

union FragU { v16bf v; bf16x8 h[2]; };
union U4   { unsigned short us[4]; uint2 u2; };

// ---------- fp32 -> bf16 split helpers (round-to-nearest-even) ----------
__device__ __forceinline__ unsigned short bf16_rne(float f) {
    unsigned u = __float_as_uint(f);
    u += 0x7fffu + ((u >> 16) & 1u);
    return (unsigned short)(u >> 16);
}
__device__ __forceinline__ float bf16_up(unsigned short h) {
    return __uint_as_float(((unsigned)h) << 16);
}
__device__ __forceinline__ void split2(float f, unsigned short& hi, unsigned short& lo) {
    unsigned short h = bf16_rne(f);
    hi = h;
    lo = bf16_rne(f - bf16_up(h));
}

// ============================================================================
// Stage 1: exact quantile(|W|, 0.9) via 4-pass MSB radix select on device.
// ws layout (uint32): [0..1] prefix (state0=rank K0, state1=rank K0+1)
//                     [2..3] remaining rank   [4] thr bits (float)
//                     [16..527] hist[2][256]
// Packed bf16 arrays (optional, if ws_size permits) start at byte 4096.
// ============================================================================
__global__ void topk_init(unsigned* ws) {
    for (int i = threadIdx.x; i < 512; i += blockDim.x) ws[16 + i] = 0u;
    if (threadIdx.x == 0) {
        ws[0] = 0u; ws[1] = 0u;
        ws[2] = (unsigned)K0; ws[3] = (unsigned)(K0 + 1);
        ws[4] = 0u;
    }
}

__global__ void topk_hist(const float* __restrict__ W, unsigned* __restrict__ ws, int shift) {
    __shared__ unsigned h[2][256];
    for (int i = threadIdx.x; i < 512; i += blockDim.x) (&h[0][0])[i] = 0u;
    __syncthreads();

    const unsigned pref0 = ws[0];
    const unsigned pref1 = ws[1];
    const unsigned prefmask = (shift == 24) ? 0u
                            : (0x7fffffffu & (0xffffffffu << (shift + 8)));

    const unsigned stride = gridDim.x * blockDim.x;
    for (unsigned i = blockIdx.x * blockDim.x + threadIdx.x; i < (unsigned)NW; i += stride) {
        unsigned bits = __float_as_uint(W[i]) & 0x7fffffffu;   // |w| bit pattern
        unsigned d = (bits >> shift) & 255u;
        if ((bits & prefmask) == pref0) atomicAdd(&h[0][d], 1u);
        if ((bits & prefmask) == pref1) atomicAdd(&h[1][d], 1u);
    }
    __syncthreads();
    for (int i = threadIdx.x; i < 512; i += blockDim.x) {
        unsigned v = (&h[0][0])[i];
        if (v) atomicAdd(&ws[16 + i], v);
    }
}

__global__ void topk_select(unsigned* ws, int shift, int final_pass) {
    const int tid = threadIdx.x;
    if (tid < 2) {
        const unsigned* h = &ws[16 + 256 * tid];
        unsigned r = ws[2 + tid];
        unsigned cum = 0, digit = 255;
        for (int b = 0; b < 256; ++b) {
            unsigned c = h[b];
            if (cum + c > r) { digit = (unsigned)b; break; }
            cum += c;
        }
        ws[2 + tid] = r - cum;
        ws[tid] |= (digit << shift);
    }
    __syncthreads();
    for (int i = tid; i < 512; i += blockDim.x) ws[16 + i] = 0u;  // reset for next pass
    __syncthreads();
    if (final_pass && tid == 0) {
        float v0 = __uint_as_float(ws[0]);   // k-th order statistic of |W|
        float v1 = __uint_as_float(ws[1]);   // (k+1)-th
        ws[4] = __float_as_uint(0.5f * (v0 + v1));   // linear-interp quantile
    }
}

// ============================================================================
// Stage 2a (packed path): one-time fp32 -> (hi,lo) bf16 split, mask fused for W.
// ============================================================================
__global__ void pack_split(const float* __restrict__ src,
                           unsigned short* __restrict__ dhi,
                           unsigned short* __restrict__ dlo,
                           unsigned n4, const unsigned* __restrict__ ws, int apply_mask) {
    const float thr = __uint_as_float(ws[4]);
    const unsigned stride = gridDim.x * blockDim.x;
    for (unsigned q = blockIdx.x * blockDim.x + threadIdx.x; q < n4; q += stride) {
        const size_t i = (size_t)q * 4;
        const float4 v = *(const float4*)(src + i);
        float f[4] = {v.x, v.y, v.z, v.w};
        U4 hi, lo;
        #pragma unroll
        for (int j = 0; j < 4; ++j) {
            float w = f[j];
            if (apply_mask) w = (__builtin_fabsf(w) >= thr) ? w : 0.0f;
            split2(w, hi.us[j], lo.us[j]);
        }
        *(uint2*)(dhi + i) = hi.u2;
        *(uint2*)(dlo + i) = lo.u2;
    }
}

// ============================================================================
// Fragment load: CDNA5 16-bit A-matrix layout.
// lane m (0..15): K 0-7 and 16-23; lane m+16: K 8-15 and 24-31.
// ============================================================================
__device__ __forceinline__ v16bf load_frag(const unsigned short* tile, int lane) {
    const int row = lane & 15;
    const int sel = (lane >> 4) & 1;
    const unsigned short* p = tile + row * LDSS + sel * 8;
    FragU f;
    f.h[0] = *(const bf16x8*)(p);
    f.h[1] = *(const bf16x8*)(p + 16);
    return f.v;
}

// gfx1250 async global -> LDS copy, 16B per lane, tracked by ASYNCcnt.
__device__ __forceinline__ void async_copy16(unsigned lds_off, const void* gptr) {
    asm volatile("global_load_async_to_lds_b128 %0, %1, off"
                 :: "v"(lds_off), "v"(gptr) : "memory");
}

// ============================================================================
// Stage 2b (packed path): bf16x3 GEMM with double-buffered async-LDS staging.
// ============================================================================
__global__ void __launch_bounds__(256, 1)
topkast_gemm_packed(const unsigned short* __restrict__ Ahi, const unsigned short* __restrict__ Alo,
                    const unsigned short* __restrict__ Whi, const unsigned short* __restrict__ Wlo,
                    const float* __restrict__ bias, float* __restrict__ C) {
    __shared__ __align__(16) unsigned short sAhi[2][BM * LDSS];
    __shared__ __align__(16) unsigned short sAlo[2][BM * LDSS];
    __shared__ __align__(16) unsigned short sBhi[2][BN * LDSS];
    __shared__ __align__(16) unsigned short sBlo[2][BN * LDSS];

    const int tid  = threadIdx.x;
    const int lane = tid & 31;
    const int wave = tid >> 5;
    const int wm   = wave & 3;            // 4 waves along M -> 32 rows each
    const int wn   = wave >> 2;           // 2 waves along N -> 64 cols each
    const int bm0  = blockIdx.y * BM;
    const int bn0  = blockIdx.x * BN;

    const unsigned baseAhi = (unsigned)(size_t)&sAhi[0][0];
    const unsigned baseAlo = (unsigned)(size_t)&sAlo[0][0];
    const unsigned baseBhi = (unsigned)(size_t)&sBhi[0][0];
    const unsigned baseBlo = (unsigned)(size_t)&sBlo[0][0];
    const unsigned stageBytes = BM * LDSS * 2;   // 10240

    v8f acc[2][4];
    const v8f vz = {0.f, 0.f, 0.f, 0.f, 0.f, 0.f, 0.f, 0.f};
    #pragma unroll
    for (int i = 0; i < 2; ++i)
        #pragma unroll
        for (int j = 0; j < 4; ++j) acc[i][j] = vz;

    // Issue 8 async b128 copies (2 per array) staging one 128x32 bf16 tile set.
    auto issue = [&](int buf, int k0) {
        const unsigned sb = buf * stageBytes;
        #pragma unroll
        for (int c = 0; c < 2; ++c) {
            const int chunk = tid + (c << 8);          // 512 chunks of 8 elems
            const int row   = chunk >> 2;
            const int col   = (chunk & 3) << 3;
            const unsigned loff = (unsigned)((row * LDSS + col) << 1);
            const size_t ga = (size_t)(bm0 + row) * IN_F + k0 + col;
            const size_t gw = (size_t)(bn0 + row) * IN_F + k0 + col;
            async_copy16(baseAhi + sb + loff, Ahi + ga);
            async_copy16(baseAlo + sb + loff, Alo + ga);
            async_copy16(baseBhi + sb + loff, Whi + gw);
            async_copy16(baseBlo + sb + loff, Wlo + gw);
        }
    };

    issue(0, 0);
    const int NK = IN_F / BKK;   // 128
    for (int i = 0; i < NK; ++i) {
        const int cur = i & 1;
        if (i + 1 < NK) {
            issue(cur ^ 1, (i + 1) * BKK);
            asm volatile("s_wait_asynccnt 0x8" ::: "memory");  // stage i done (in-order)
        } else {
            asm volatile("s_wait_asynccnt 0x0" ::: "memory");
        }
        __syncthreads();

        v16bf ah[2], al[2], bh[4], bl[4];
        #pragma unroll
        for (int mi = 0; mi < 2; ++mi) {
            const int r0 = (wm * 32 + mi * 16) * LDSS;
            ah[mi] = load_frag(&sAhi[cur][r0], lane);
            al[mi] = load_frag(&sAlo[cur][r0], lane);
        }
        #pragma unroll
        for (int ni = 0; ni < 4; ++ni) {
            const int r0 = (wn * 64 + ni * 16) * LDSS;
            bh[ni] = load_frag(&sBhi[cur][r0], lane);
            bl[ni] = load_frag(&sBlo[cur][r0], lane);
        }
        #pragma unroll
        for (int mi = 0; mi < 2; ++mi) {
            #pragma unroll
            for (int ni = 0; ni < 4; ++ni) {
                acc[mi][ni] = __builtin_amdgcn_wmma_f32_16x16x32_bf16(
                    false, ah[mi], false, bh[ni], (short)0, acc[mi][ni], false, false);
                acc[mi][ni] = __builtin_amdgcn_wmma_f32_16x16x32_bf16(
                    false, ah[mi], false, bl[ni], (short)0, acc[mi][ni], false, false);
                acc[mi][ni] = __builtin_amdgcn_wmma_f32_16x16x32_bf16(
                    false, al[mi], false, bh[ni], (short)0, acc[mi][ni], false, false);
            }
        }
        __syncthreads();   // all waves done reading buffer before it is re-staged
    }

    // ---- epilogue: C/D layout = lane%16 -> N, vgpr r -> M=r (+8 for upper half-wave) ----
    const int colInTile = lane & 15;
    const int rowOff    = (lane >> 4) * 8;
    #pragma unroll
    for (int mi = 0; mi < 2; ++mi) {
        #pragma unroll
        for (int ni = 0; ni < 4; ++ni) {
            const int gm0 = bm0 + wm * 32 + mi * 16 + rowOff;
            const int gn  = bn0 + wn * 64 + ni * 16 + colInTile;
            const float bv = bias[gn];
            #pragma unroll
            for (int r = 0; r < 8; ++r) {
                C[(size_t)(gm0 + r) * OUT_F + gn] = acc[mi][ni][r] + bv;
            }
        }
    }
}

// ============================================================================
// Stage 2 (fallback, small ws): fused-conversion bf16x3 GEMM (round-1 version).
// ============================================================================
__global__ void __launch_bounds__(256, 2)
topkast_gemm_fused(const float* __restrict__ A, const float* __restrict__ W,
                   const float* __restrict__ bias, const unsigned* __restrict__ ws,
                   float* __restrict__ C) {
    __shared__ __align__(16) unsigned short sAhi[BM * LDSS];
    __shared__ __align__(16) unsigned short sAlo[BM * LDSS];
    __shared__ __align__(16) unsigned short sBhi[BN * LDSS];
    __shared__ __align__(16) unsigned short sBlo[BN * LDSS];

    const int tid  = threadIdx.x;
    const int lane = tid & 31;
    const int wave = tid >> 5;
    const int wm   = wave & 3;
    const int wn   = wave >> 2;
    const int bm0  = blockIdx.y * BM;
    const int bn0  = blockIdx.x * BN;

    const float thr = __uint_as_float(ws[4]);

    v8f acc[2][4];
    const v8f vz = {0.f, 0.f, 0.f, 0.f, 0.f, 0.f, 0.f, 0.f};
    #pragma unroll
    for (int i = 0; i < 2; ++i)
        #pragma unroll
        for (int j = 0; j < 4; ++j) acc[i][j] = vz;

    for (int k0 = 0; k0 < IN_F; k0 += BKK) {
        #pragma unroll
        for (int c = 0; c < 4; ++c) {
            const int chunk = tid + c * 256;
            const int row   = chunk >> 3;
            const int col   = (chunk & 7) << 2;
            const size_t aoff = (size_t)(bm0 + row) * IN_F + k0 + col;
            const size_t woff = (size_t)(bn0 + row) * IN_F + k0 + col;

            const float4 va = *(const float4*)(&A[aoff]);
            const float4 vw = *(const float4*)(&W[woff]);
            if (k0 + BKK < IN_F) {
                __builtin_prefetch(&A[aoff + BKK], 0, 1);
                __builtin_prefetch(&W[woff + BKK], 0, 1);
            }

            float af[4] = {va.x, va.y, va.z, va.w};
            float wf[4] = {vw.x, vw.y, vw.z, vw.w};
            #pragma unroll
            for (int j = 0; j < 4; ++j) {
                unsigned short hi, lo;
                split2(af[j], hi, lo);
                sAhi[row * LDSS + col + j] = hi;
                sAlo[row * LDSS + col + j] = lo;
                float w = (__builtin_fabsf(wf[j]) >= thr) ? wf[j] : 0.0f;
                split2(w, hi, lo);
                sBhi[row * LDSS + col + j] = hi;
                sBlo[row * LDSS + col + j] = lo;
            }
        }
        __syncthreads();

        v16bf ah[2], al[2], bh[4], bl[4];
        #pragma unroll
        for (int mi = 0; mi < 2; ++mi) {
            const int r0 = (wm * 32 + mi * 16) * LDSS;
            ah[mi] = load_frag(&sAhi[r0], lane);
            al[mi] = load_frag(&sAlo[r0], lane);
        }
        #pragma unroll
        for (int ni = 0; ni < 4; ++ni) {
            const int r0 = (wn * 64 + ni * 16) * LDSS;
            bh[ni] = load_frag(&sBhi[r0], lane);
            bl[ni] = load_frag(&sBlo[r0], lane);
        }
        #pragma unroll
        for (int mi = 0; mi < 2; ++mi) {
            #pragma unroll
            for (int ni = 0; ni < 4; ++ni) {
                acc[mi][ni] = __builtin_amdgcn_wmma_f32_16x16x32_bf16(
                    false, ah[mi], false, bh[ni], (short)0, acc[mi][ni], false, false);
                acc[mi][ni] = __builtin_amdgcn_wmma_f32_16x16x32_bf16(
                    false, ah[mi], false, bl[ni], (short)0, acc[mi][ni], false, false);
                acc[mi][ni] = __builtin_amdgcn_wmma_f32_16x16x32_bf16(
                    false, al[mi], false, bh[ni], (short)0, acc[mi][ni], false, false);
            }
        }
        __syncthreads();
    }

    const int colInTile = lane & 15;
    const int rowOff    = (lane >> 4) * 8;
    #pragma unroll
    for (int mi = 0; mi < 2; ++mi) {
        #pragma unroll
        for (int ni = 0; ni < 4; ++ni) {
            const int gm0 = bm0 + wm * 32 + mi * 16 + rowOff;
            const int gn  = bn0 + wn * 64 + ni * 16 + colInTile;
            const float bv = bias[gn];
            #pragma unroll
            for (int r = 0; r < 8; ++r) {
                C[(size_t)(gm0 + r) * OUT_F + gn] = acc[mi][ni][r] + bv;
            }
        }
    }
}

// ============================================================================
extern "C" void kernel_launch(void* const* d_in, const int* in_sizes, int n_in,
                              void* d_out, int out_size, void* d_ws, size_t ws_size,
                              hipStream_t stream) {
    const float* A    = (const float*)d_in[0];   // inputs [2048,4096] f32
    const float* W    = (const float*)d_in[1];   // weight [4096,4096] f32
    const float* bias = (const float*)d_in[2];   // bias   [4096]      f32
    float* out  = (float*)d_out;                 // [2048,4096] f32
    unsigned* ws = (unsigned*)d_ws;

    // ---- threshold (always) ----
    topk_init<<<1, 256, 0, stream>>>(ws);
    const int shifts[4] = {24, 16, 8, 0};
    for (int p = 0; p < 4; ++p) {
        topk_hist<<<1024, 256, 0, stream>>>(W, ws, shifts[p]);
        topk_select<<<1, 256, 0, stream>>>(ws, shifts[p], p == 3);
    }

    dim3 grid(OUT_F / BN, BSZ / BM);   // 32 x 16 blocks

    const size_t nA = (size_t)BSZ * IN_F;        // 8.4M
    const size_t nW = (size_t)OUT_F * IN_F;      // 16.8M
    const size_t needed = 4096 + (nA + nW) * 2 * sizeof(unsigned short);  // ~101 MB

    if (ws_size >= needed) {
        unsigned short* Ahi = (unsigned short*)((char*)d_ws + 4096);
        unsigned short* Alo = Ahi + nA;
        unsigned short* Whi = Alo + nA;
        unsigned short* Wlo = Whi + nW;
        pack_split<<<8192, 256, 0, stream>>>(A, Ahi, Alo, (unsigned)(nA / 4), ws, 0);
        pack_split<<<16384, 256, 0, stream>>>(W, Whi, Wlo, (unsigned)(nW / 4), ws, 1);
        topkast_gemm_packed<<<grid, 256, 0, stream>>>(Ahi, Alo, Whi, Wlo, bias, out);
    } else {
        topkast_gemm_fused<<<grid, 256, 0, stream>>>(A, W, bias, ws, out);
    }
    (void)in_sizes; (void)n_in; (void)out_size;
}
